// Model_31095563223583
// MI455X (gfx1250) — matrix operand order repaired and checked by reading, not compiled, not run on hardware
//
#include <hip/hip_runtime.h>
#include <hip/hip_bf16.h>

// ---------------- problem constants ----------------
#define NFIELD   20
#define FDIM     50000
#define FEATN    1000000          // NFIELD * FDIM
#define EMB      64
#define HID      256
#define BATCH    4096
#define INST     (BATCH * NFIELD) // 81920 feature instances
#define EPS_BN   1e-5f

// padded LDS row strides (halves): 16B-aligned rows, 36*n mod 64 covers all banks
#define AK64     72
#define BK64     72
#define AK256    264
#define BK256    264

typedef __attribute__((ext_vector_type(16))) _Float16 v16h;
typedef __attribute__((ext_vector_type(8)))  _Float16 v8h;
typedef __attribute__((ext_vector_type(8)))  float    v8f;

// ---------------- workspace layout (bytes) ----------------
static constexpr size_t OFF_CNT     = 0;                                   // int [FEATN]
static constexpr size_t OFF_MAP     = OFF_CNT     + (size_t)FEATN * 4;     // int [FEATN]
static constexpr size_t OFF_NSLOT   = OFF_MAP     + (size_t)FEATN * 4;     // int + pad
static constexpr size_t OFF_SDINV   = OFF_NSLOT   + 16;                    // float (1/sqrt(deg_s))
static constexpr size_t OFF_SLOTAGG = OFF_NSLOT   + 256;                   // f32 [INST][EMB]
static constexpr size_t OFF_UBUF    = OFF_SLOTAGG + (size_t)INST * EMB * 4;// f32 [INST][EMB]
static constexpr size_t OFF_VS      = OFF_UBUF    + (size_t)INST * EMB * 4;// f32 [BATCH][EMB]
static constexpr size_t OFF_EBUF    = OFF_VS      + (size_t)BATCH * EMB * 4;//f32 [INST][EMB]
static constexpr size_t OFF_HSUM    = OFF_EBUF    + (size_t)INST * EMB * 4;// f32 [BATCH][EMB]
static constexpr size_t OFF_FMLIN   = OFF_HSUM    + (size_t)BATCH * EMB * 4;//f32 [BATCH]
static constexpr size_t OFF_Z1      = OFF_FMLIN   + (size_t)BATCH * 4;     // f32 [BATCH][HID]
static constexpr size_t OFF_Z2      = OFF_Z1      + (size_t)BATCH * HID * 4;
static constexpr size_t OFF_MEAN    = OFF_Z2      + (size_t)BATCH * HID * 4;
static constexpr size_t OFF_RSTD    = OFF_MEAN    + (size_t)HID * 4;
// f16 transposed weight copies ([N][K], packed) for async LDS staging
static constexpr size_t OFF_W0T     = OFF_RSTD + (size_t)HID * 4;          // [64][64] f16
static constexpr size_t OFF_W1T     = OFF_W0T  + 64 * 64 * 2;              // [64][64] f16
static constexpr size_t OFF_M1T     = OFF_W1T  + 64 * 64 * 2;              // [256][64] f16
static constexpr size_t OFF_M2T     = OFF_M1T  + 256 * 64 * 2;             // [256][256] f16
static constexpr size_t ZERO_WORDS  = OFF_UBUF / 4;  // cnt+map+nslot+slotAgg zeroed per call

// ---------------- async global->LDS staging (gfx1250) ----------------
__device__ __forceinline__ void async_chunk16(unsigned int lds_off, const void* gp) {
  asm volatile("global_load_async_to_lds_b128 %0, %1, off"
               :: "v"(lds_off), "v"(gp) : "memory");
}
__device__ __forceinline__ void async_wait0() {
  asm volatile("s_wait_asynccnt 0x0" ::: "memory");
}

// copy [NROWS][K] packed f16 (global) into LDS rows with stride LDB, 16B per lane-op
template <int NROWS, int K, int LDB>
__device__ __forceinline__ void stage_bt(const _Float16* __restrict__ g, _Float16* lds) {
  constexpr int CH = NROWS * (K / 8);
  for (int t = threadIdx.x; t < CH; t += blockDim.x) {
    int n = t / (K / 8), kc = t % (K / 8);
    async_chunk16((unsigned int)(unsigned long long)(lds + n * LDB + kc * 8),
                  (const void*)(g + (size_t)n * K + kc * 8));
  }
}

// ---------------- WMMA helpers (wave32, 16x16 tile) ----------------
__device__ __forceinline__ v8h ldsv8(const _Float16* p) { return *(const v8h*)p; }
__device__ __forceinline__ v16h cat8(v8h lo, v8h hi) {
  return __builtin_shufflevector(lo, hi, 0, 1, 2, 3, 4, 5, 6, 7,
                                 8, 9, 10, 11, 12, 13, 14, 15);
}

// A: row-major [16][lda] f16 in LDS; Bt: B transposed, rows are N: [16][ldb] f16 in LDS.
// One v_wmma_f32_16x16x32_f16 covering K = k0..k0+31.
__device__ __forceinline__ v8f wmma_step(const _Float16* A, int lda,
                                         const _Float16* Bt, int ldb,
                                         int k0, v8f c) {
  const int lane = threadIdx.x & 31;
  const int mn = lane & 15;
  const int hi = (lane & 16) ? 1 : 0;
  const int kA = k0 + hi * 8;    // lanes 0-15: K 0..7 & 16..23 ; lanes 16-31: +8
  const int kB = k0 + hi * 8;    // lanes 0-15: K 0..7 & 16..23 ; lanes 16-31: +8
  v16h a = cat8(ldsv8(A + mn * lda + kA), ldsv8(A + mn * lda + kA + 16));
  v16h b = cat8(ldsv8(Bt + mn * ldb + kB), ldsv8(Bt + mn * ldb + kB + 16));
  return __builtin_amdgcn_wmma_f32_16x16x32_f16(false, a, false, b, (short)0, c, false, false);
}

__device__ __forceinline__ v8f wmma_k64(const _Float16* A, int lda,
                                        const _Float16* Bt, int ldb, v8f c) {
  c = wmma_step(A, lda, Bt, ldb, 0, c);
  c = wmma_step(A, lda, Bt, ldb, 32, c);
  return c;
}

__device__ __forceinline__ v8f wmma_k256(const _Float16* A, int lda,
                                         const _Float16* Bt, int ldb, v8f c) {
#pragma unroll
  for (int k0 = 0; k0 < 256; k0 += 32) c = wmma_step(A, lda, Bt, ldb, k0, c);
  return c;
}

__device__ __forceinline__ float sample_dinv(const unsigned char* fmask) {
  int c = 1;
#pragma unroll
  for (int j = 0; j < NFIELD; ++j) c += fmask[j] ? 1 : 0;
  return rsqrtf((float)c);
}

// ---------------- kernels ----------------
__global__ void k_zero(unsigned int* p, size_t n) {
  size_t i = (size_t)blockIdx.x * blockDim.x + threadIdx.x;
  size_t stride = (size_t)gridDim.x * blockDim.x;
  for (; i < n; i += stride) p[i] = 0u;
}

// f32 [K][N] -> f16 transposed [N][K] (packed) weight copies for async staging
__global__ void k_cvtT(const float* __restrict__ src, _Float16* __restrict__ dst,
                       int K, int N) {
  int idx = blockIdx.x * blockDim.x + threadIdx.x;
  if (idx >= K * N) return;
  int n = idx / K, k = idx - n * K;
  dst[idx] = (_Float16)src[(size_t)k * N + n];
}

// degree counting + compact slot assignment for unique in-batch features
__global__ void k_count(const int* __restrict__ x, const unsigned char* __restrict__ fmask,
                        int* cnt, int* map, int* nslots, float* sdinv) {
  int i = blockIdx.x * blockDim.x + threadIdx.x;
  if (i == 0) *sdinv = sample_dinv(fmask);
  if (i >= INST) return;
  int j = i % NFIELD;
  if (!fmask[j]) return;
  int f = x[i] + j * FDIM;
  atomicAdd(&cnt[f], 1);
  if (atomicCAS(&map[f], 0, 1) == 0) {
    int s = atomicAdd(nslots, 1);
    atomicExch(&map[f], s + 2);   // slot = map[f]-2, resolved before any reader kernel
  }
}

// U[i] = w_masked[f_i] @ W0   (4 waves x one 16-row tile each)
__global__ void k_u(const int* __restrict__ x, const unsigned char* __restrict__ fmask,
                    const unsigned char* __restrict__ kmask,
                    const float* __restrict__ w, const _Float16* __restrict__ W0t,
                    float* __restrict__ Ubuf) {
  __shared__ __align__(16) _Float16 Bs[64 * BK64];
  __shared__ __align__(16) _Float16 As[4][16 * AK64];
  __shared__ int rowF[64];
  const int tid = threadIdx.x;
  stage_bt<64, 64, BK64>(W0t, Bs);                      // async B tile
  const int tile0 = blockIdx.x * 4;
  if (tid < 64) {
    int i = tile0 * 16 + tid;
    int j = i % NFIELD;
    int f = x[i] + j * FDIM;
    rowF[tid] = (fmask[j] && kmask[j]) ? f : -1;
  }
  __syncthreads();
  const int wave = tid >> 5, lane = tid & 31;
  for (int t = lane; t < 16 * 64; t += 32) {
    int r = t >> 6, d = t & 63;
    int f = rowF[wave * 16 + r];
    As[wave][r * AK64 + d] = (f >= 0) ? (_Float16)w[(size_t)f * 64 + d] : (_Float16)0.f;
  }
  async_wait0();
  __syncthreads();
  const int ti = (tile0 + wave) * 16;
  const int n = lane & 15, mb = (lane & 16) ? 8 : 0;
  for (int n0 = 0; n0 < 64; n0 += 16) {
    v8f c = {};
    c = wmma_k64(&As[wave][0], AK64, &Bs[n0 * BK64], BK64, c);
#pragma unroll
    for (int r = 0; r < 8; ++r)
      Ubuf[(size_t)(ti + mb + r) * 64 + n0 + n] = c[r];
  }
}

// h1_s = b0 + dinv_s * sum_j dinv_f * U[s,j] ; v_s = h1_s @ W1
__global__ void k_h1vs(const int* __restrict__ x, const unsigned char* __restrict__ fmask,
                       const _Float16* __restrict__ W1t, const float* __restrict__ b0,
                       const int* __restrict__ cnt, const float* __restrict__ sdinv,
                       const float* __restrict__ Ubuf, float* __restrict__ vs) {
  __shared__ __align__(16) _Float16 Bs[64 * BK64];
  __shared__ __align__(16) _Float16 As[4][16 * AK64];
  __shared__ float dvf[4][16 * NFIELD];
  const int tid = threadIdx.x;
  stage_bt<64, 64, BK64>(W1t, Bs);
  const int wave = tid >> 5, lane = tid & 31;
  const int s0 = (blockIdx.x * 4 + wave) * 16;
  const float dinv_s = *sdinv;
  for (int t = lane; t < 16 * NFIELD; t += 32) {
    int r = t / NFIELD, j = t % NFIELD;
    int f = x[(s0 + r) * NFIELD + j] + j * FDIM;
    dvf[wave][t] = fmask[j] ? rsqrtf((float)cnt[f] + 1.f) : 0.f;
  }
  __syncthreads();
  for (int t = lane; t < 16 * 64; t += 32) {
    int r = t >> 6, d = t & 63;
    float acc = 0.f;
#pragma unroll
    for (int j = 0; j < NFIELD; ++j)
      acc += dvf[wave][r * NFIELD + j] * Ubuf[((size_t)(s0 + r) * NFIELD + j) * 64 + d];
    As[wave][r * AK64 + d] = (_Float16)(b0[d] + dinv_s * acc);
  }
  async_wait0();
  __syncthreads();
  const int n = lane & 15, mb = (lane & 16) ? 8 : 0;
  for (int n0 = 0; n0 < 64; n0 += 16) {
    v8f c = {};
    c = wmma_k64(&As[wave][0], AK64, &Bs[n0 * BK64], BK64, c);
#pragma unroll
    for (int r = 0; r < 8; ++r)
      vs[(size_t)(s0 + mb + r) * 64 + n0 + n] = c[r];
  }
}

// slotAgg[slot(f)] += dinv_s * v_s[s]  for every instance edge (s -> f)
__global__ void k_scatter(const int* __restrict__ x, const unsigned char* __restrict__ fmask,
                          const int* __restrict__ map, const float* __restrict__ sdinv,
                          const float* __restrict__ vs, float* __restrict__ slotAgg) {
  int idx = blockIdx.x * blockDim.x + threadIdx.x;
  if (idx >= INST * EMB) return;
  int i = idx >> 6, d = idx & 63;
  int j = i % NFIELD;
  if (!fmask[j]) return;
  int f = x[i] + j * FDIM;
  int slot = map[f] - 2;
  atomicAdd(&slotAgg[(size_t)slot * 64 + d], (*sdinv) * vs[(size_t)(i / NFIELD) * 64 + d]);
}

// v_f = (U*dinv_f^2 + b0) @ W1 ; Ebuf[i] = fmask * (dinv_f*slotAgg + v_f*dinv_f^2 + b1)
__global__ void k_emb(const int* __restrict__ x, const unsigned char* __restrict__ fmask,
                      const _Float16* __restrict__ W1t, const float* __restrict__ b0,
                      const float* __restrict__ b1, const int* __restrict__ cnt,
                      const int* __restrict__ map, const float* __restrict__ Ubuf,
                      const float* __restrict__ slotAgg, float* __restrict__ Ebuf) {
  __shared__ __align__(16) _Float16 Bs[64 * BK64];
  __shared__ __align__(16) _Float16 As[4][16 * AK64];
  __shared__ float rdv[4][16];
  __shared__ int   rsl[4][16];
  const int tid = threadIdx.x, wave = tid >> 5, lane = tid & 31;
  stage_bt<64, 64, BK64>(W1t, Bs);
  const int i0 = (blockIdx.x * 4 + wave) * 16;
  if (lane < 16) {
    int i = i0 + lane;
    int j = i % NFIELD;
    int f = x[i] + j * FDIM;
    bool v = fmask[j];
    rdv[wave][lane] = v ? rsqrtf((float)cnt[f] + 1.f) : 0.f;
    rsl[wave][lane] = v ? (map[f] - 2) : -1;
  }
  __syncthreads();
  for (int t = lane; t < 16 * 64; t += 32) {
    int r = t >> 6, d = t & 63;
    float dv = rdv[wave][r];
    As[wave][r * AK64 + d] = (_Float16)(Ubuf[(size_t)(i0 + r) * 64 + d] * dv * dv + b0[d]);
  }
  async_wait0();
  __syncthreads();
  const int n = lane & 15, mb = (lane & 16) ? 8 : 0;
  for (int n0 = 0; n0 < 64; n0 += 16) {
    v8f c = {};
    c = wmma_k64(&As[wave][0], AK64, &Bs[n0 * BK64], BK64, c);
    float bb = b1[n0 + n];
#pragma unroll
    for (int r = 0; r < 8; ++r) {
      int rr = mb + r;
      int slot = rsl[wave][rr];
      float dv = rdv[wave][rr];
      float val = (slot >= 0)
        ? dv * slotAgg[(size_t)slot * 64 + n0 + n] + c[r] * dv * dv + bb
        : 0.f;
      Ebuf[(size_t)(i0 + rr) * 64 + n0 + n] = val;
    }
  }
}

// per-sample: hsum, FM interaction, linear term   (4 samples/block, 64 thr/sample)
__global__ void k_sample(const int* __restrict__ x, const unsigned char* __restrict__ fmask,
                         const float* __restrict__ bvec, const float* __restrict__ w,
                         const float* __restrict__ Ebuf, float* __restrict__ hsum,
                         float* __restrict__ fmlin) {
  __shared__ float red[256];
  const int tid = threadIdx.x;
  const int sub = tid >> 6, d = tid & 63;
  const int s = blockIdx.x * 4 + sub;
  float se = 0.f, se_ = 0.f, dot = 0.f, lin = 0.f;
#pragma unroll
  for (int j = 0; j < NFIELD; ++j) {
    int i = s * NFIELD + j;
    int f = x[i] + j * FDIM;
    float e  = fmask[j] ? w[(size_t)f * 64 + d] : 0.f;
    float e_ = Ebuf[(size_t)i * 64 + d];
    se += e; se_ += e_; dot += e * e_;
    if (d == 0) lin += bvec[f];
  }
  hsum[(size_t)s * 64 + d] = se_;
  red[tid] = se * se_ - dot;
  __syncthreads();
  for (int off = 32; off >= 1; off >>= 1) {
    if (d < off) red[tid] += red[tid + off];
    __syncthreads();
  }
  if (d == 0) fmlin[s] = lin + 0.5f * red[tid];
}

// [BATCH][64] @ W1 (via W1^T f16 [256][64]) + bias -> out[BATCH][HID]
__global__ void k_gemm64(const float* __restrict__ A, const _Float16* __restrict__ Bt,
                         const float* __restrict__ bias, float* __restrict__ out) {
  __shared__ __align__(16) _Float16 As[16 * AK64];
  __shared__ __align__(16) _Float16 Bs[64 * BK64];
  const int tid = threadIdx.x;
  const int m0 = (blockIdx.x >> 2) * 16;
  const int cb = (blockIdx.x & 3) * 64;
  stage_bt<64, 64, BK64>(Bt + (size_t)cb * 64, Bs);   // rows n = cb..cb+63
  for (int t = tid; t < 16 * 64; t += 128)
    As[(t >> 6) * AK64 + (t & 63)] = (_Float16)A[(size_t)(m0 + (t >> 6)) * 64 + (t & 63)];
  async_wait0();
  __syncthreads();
  const int wave = tid >> 5, lane = tid & 31;
  const int n = lane & 15, mb = (lane & 16) ? 8 : 0;
  const int c0 = wave * 16;
  v8f c = {};
  c = wmma_k64(As, AK64, &Bs[c0 * BK64], BK64, c);
  float bb = bias[cb + c0 + n];
#pragma unroll
  for (int r = 0; r < 8; ++r)
    out[(size_t)(m0 + mb + r) * HID + cb + c0 + n] = c[r] + bb;
}

// [BATCH][256] @ W2 (via W2^T f16 [256][256]) + bias -> out[BATCH][HID]
__global__ void k_gemm256(const float* __restrict__ A, const _Float16* __restrict__ Bt,
                          const float* __restrict__ bias, float* __restrict__ out) {
  __shared__ __align__(16) _Float16 As[16 * AK256];
  __shared__ __align__(16) _Float16 Bs[64 * BK256];
  const int tid = threadIdx.x;
  const int m0 = (blockIdx.x >> 2) * 16;
  const int cb = (blockIdx.x & 3) * 64;
  stage_bt<64, 256, BK256>(Bt + (size_t)cb * 256, Bs);
  for (int t = tid; t < 16 * 256; t += 128)
    As[(t >> 8) * AK256 + (t & 255)] = (_Float16)A[(size_t)(m0 + (t >> 8)) * HID + (t & 255)];
  async_wait0();
  __syncthreads();
  const int wave = tid >> 5, lane = tid & 31;
  const int n = lane & 15, mb = (lane & 16) ? 8 : 0;
  const int c0 = wave * 16;
  v8f c = {};
  c = wmma_k256(As, AK256, &Bs[c0 * BK256], BK256, c);
  float bb = bias[cb + c0 + n];
#pragma unroll
  for (int r = 0; r < 8; ++r)
    out[(size_t)(m0 + mb + r) * HID + cb + c0 + n] = c[r] + bb;
}

// batchnorm stats per column (train mode: biased variance over batch)
__global__ void k_bnstat(const float* __restrict__ z, float* __restrict__ mean,
                         float* __restrict__ rstd) {
  __shared__ float s1[256], s2[256];
  const int col = blockIdx.x, tid = threadIdx.x;
  float a = 0.f, b = 0.f;
  for (int r = tid; r < BATCH; r += 256) {
    float v = z[(size_t)r * HID + col];
    a += v; b += v * v;
  }
  s1[tid] = a; s2[tid] = b;
  __syncthreads();
  for (int off = 128; off >= 1; off >>= 1) {
    if (tid < off) { s1[tid] += s1[tid + off]; s2[tid] += s2[tid + off]; }
    __syncthreads();
  }
  if (tid == 0) {
    float m = s1[0] / (float)BATCH;
    float var = s2[0] / (float)BATCH - m * m;
    mean[col] = m;
    rstd[col] = rsqrtf(var + EPS_BN);
  }
}

__global__ void k_bnrelu(float* __restrict__ z, const float* __restrict__ mean,
                         const float* __restrict__ rstd, const float* __restrict__ g,
                         const float* __restrict__ beta) {
  int idx = blockIdx.x * blockDim.x + threadIdx.x;
  if (idx >= BATCH * HID) return;
  int col = idx & (HID - 1);
  float v = (z[idx] - mean[col]) * rstd[col] * g[col] + beta[col];
  z[idx] = v > 0.f ? v : 0.f;
}

__global__ void k_final(const float* __restrict__ z2, const float* __restrict__ W3,
                        const float* __restrict__ b3, const float* __restrict__ fmlin,
                        float* __restrict__ out) {
  int s = blockIdx.x * blockDim.x + threadIdx.x;
  if (s >= BATCH) return;
  float acc = 0.f;
#pragma unroll 8
  for (int k = 0; k < HID; ++k) acc += z2[(size_t)s * HID + k] * W3[k];
  out[s] = acc + b3[0] + fmlin[s];
}

// ---------------- launcher ----------------
extern "C" void kernel_launch(void* const* d_in, const int* in_sizes, int n_in,
                              void* d_out, int out_size, void* d_ws, size_t ws_size,
                              hipStream_t stream) {
  const int*           x      = (const int*)d_in[0];
  const unsigned char* fmask  = (const unsigned char*)d_in[1];
  const unsigned char* kmask  = (const unsigned char*)d_in[2];
  const float*         bvec   = (const float*)d_in[3];
  const float*         w      = (const float*)d_in[4];
  const float*         gcn_W  = (const float*)d_in[5];   // [2][64][64]
  const float*         gcn_b  = (const float*)d_in[6];   // [2][64]
  const float*         mlp_W1 = (const float*)d_in[7];
  const float*         mlp_b1 = (const float*)d_in[8];
  const float*         bn1_g  = (const float*)d_in[9];
  const float*         bn1_b  = (const float*)d_in[10];
  const float*         mlp_W2 = (const float*)d_in[11];
  const float*         mlp_b2 = (const float*)d_in[12];
  const float*         bn2_g  = (const float*)d_in[13];
  const float*         bn2_b  = (const float*)d_in[14];
  const float*         mlp_W3 = (const float*)d_in[15];
  const float*         mlp_b3 = (const float*)d_in[16];
  float*               out    = (float*)d_out;

  char* ws = (char*)d_ws;
  int*      cnt     = (int*)(ws + OFF_CNT);
  int*      map     = (int*)(ws + OFF_MAP);
  int*      nslots  = (int*)(ws + OFF_NSLOT);
  float*    sdinv   = (float*)(ws + OFF_SDINV);
  float*    slotAgg = (float*)(ws + OFF_SLOTAGG);
  float*    Ubuf    = (float*)(ws + OFF_UBUF);
  float*    vs      = (float*)(ws + OFF_VS);
  float*    Ebuf    = (float*)(ws + OFF_EBUF);
  float*    hsum    = (float*)(ws + OFF_HSUM);
  float*    fmlin   = (float*)(ws + OFF_FMLIN);
  float*    z1      = (float*)(ws + OFF_Z1);
  float*    z2      = (float*)(ws + OFF_Z2);
  float*    mean    = (float*)(ws + OFF_MEAN);
  float*    rstd    = (float*)(ws + OFF_RSTD);
  _Float16* W0t     = (_Float16*)(ws + OFF_W0T);
  _Float16* W1t     = (_Float16*)(ws + OFF_W1T);
  _Float16* M1t     = (_Float16*)(ws + OFF_M1T);
  _Float16* M2t     = (_Float16*)(ws + OFF_M2T);

  const float* b0 = gcn_b;              // layer 0 bias
  const float* b1 = gcn_b + 64;         // layer 1 bias

  // 1) zero counters / map / slot accumulators; build f16 transposed weights
  k_zero<<<2048, 256, 0, stream>>>((unsigned int*)ws, ZERO_WORDS);
  k_cvtT<<<(64 * 64 + 255) / 256, 256, 0, stream>>>(gcn_W, W0t, 64, 64);
  k_cvtT<<<(64 * 64 + 255) / 256, 256, 0, stream>>>(gcn_W + 64 * 64, W1t, 64, 64);
  k_cvtT<<<(64 * 256 + 255) / 256, 256, 0, stream>>>(mlp_W1, M1t, 64, 256);
  k_cvtT<<<(256 * 256 + 255) / 256, 256, 0, stream>>>(mlp_W2, M2t, 256, 256);
  // 2) degree count + slot assignment
  k_count<<<(INST + 255) / 256, 256, 0, stream>>>(x, fmask, cnt, map, nslots, sdinv);
  // 3) U = w_masked @ W0 per instance
  k_u<<<INST / 64, 128, 0, stream>>>(x, fmask, kmask, w, W0t, Ubuf);
  // 4) sample-node layer-1 state and v_s = h1_s @ W1
  k_h1vs<<<BATCH / 64, 128, 0, stream>>>(x, fmask, W1t, b0, cnt, sdinv, Ubuf, vs);
  // 5) scatter sample contributions into per-unique-feature slots
  k_scatter<<<(INST * EMB) / 256, 256, 0, stream>>>(x, fmask, map, sdinv, vs, slotAgg);
  // 6) v_f and final feature embedding (x_emb_) per instance
  k_emb<<<INST / 64, 128, 0, stream>>>(x, fmask, W1t, b0, b1, cnt, map, Ubuf, slotAgg, Ebuf);
  // 7) FM interaction + hsum + linear term
  k_sample<<<BATCH / 4, 256, 0, stream>>>(x, fmask, bvec, w, Ebuf, hsum, fmlin);
  // 8) MLP layer 1: hsum @ W1 + b1
  k_gemm64<<<(BATCH / 16) * 4, 128, 0, stream>>>(hsum, M1t, mlp_b1, z1);
  // 9) BN1 + ReLU
  k_bnstat<<<HID, 256, 0, stream>>>(z1, mean, rstd);
  k_bnrelu<<<(BATCH * HID) / 256, 256, 0, stream>>>(z1, mean, rstd, bn1_g, bn1_b);
  // 10) MLP layer 2: z1 @ W2 + b2
  k_gemm256<<<(BATCH / 16) * 4, 128, 0, stream>>>(z1, M2t, mlp_b2, z2);
  // 11) BN2 + ReLU
  k_bnstat<<<HID, 256, 0, stream>>>(z2, mean, rstd);
  k_bnrelu<<<(BATCH * HID) / 256, 256, 0, stream>>>(z2, mean, rstd, bn2_g, bn2_b);
  // 12) final projection + combine with linear + FM terms
  k_final<<<(BATCH + 255) / 256, 256, 0, stream>>>(z2, mlp_W3, mlp_b3, fmlin, out);
}
